// SelfAttention_3573412790631
// MI455X (gfx1250) — compile-verified, matrix operand order checked by reading, not run on hardware
//
#include <hip/hip_runtime.h>
#include <stdint.h>

#define B_  4
#define C_  64
#define N_  4096
#define C8_ 8

#define NSEG_ 4
#define SEGJ_ (N_ / NSEG_)          // 1024 j per stats segment
#define CH_   256                    // K rows staged in LDS per chunk

typedef __attribute__((ext_vector_type(16))) __bf16 v16bf;
typedef __attribute__((ext_vector_type(8)))  float  v8f;

union V16 { unsigned u[8]; v16bf v; };

#define LOG2E_ 1.4426950408889634f

__device__ inline unsigned short f2bf(float f) {
    unsigned u = __float_as_uint(f);
    unsigned r = u + 0x7FFFu + ((u >> 16) & 1u);   // round-to-nearest-even
    return (unsigned short)(r >> 16);
}
__device__ inline float bf2f(unsigned short h) {
    return __uint_as_float(((unsigned)h) << 16);
}

// ---------------------------------------------------------------------------
// Kernel 1: fused QKV projection.  x [B][64][N] f32 -> Qp,Kp [B][N][16] bf16
// (32 B rows; elements 8..15 zeroed = WMMA K-padding, so attn-kernel lanes can
// load unconditionally), V [B][64][N] bf16, biases fused.
// ---------------------------------------------------------------------------
__global__ __launch_bounds__(256) void proj_kernel(
    const float* __restrict__ x,
    const float* __restrict__ wq, const float* __restrict__ bq,
    const float* __restrict__ wk, const float* __restrict__ bk,
    const float* __restrict__ wv, const float* __restrict__ bv,
    uint4* __restrict__ Qp, uint4* __restrict__ Kp,
    unsigned short* __restrict__ Vb)
{
    __shared__ float swq[C8_ * C_], swk[C8_ * C_], swv[C_ * C_];
    __shared__ float sbq[C8_], sbk[C8_], sbv[C_];
    int t = threadIdx.x;
    for (int i = t; i < C8_ * C_; i += 256) { swq[i] = wq[i]; swk[i] = wk[i]; }
    for (int i = t; i < C_ * C_;  i += 256) swv[i] = wv[i];
    if (t < C8_) { sbq[t] = bq[t]; sbk[t] = bk[t]; }
    if (t < C_)  sbv[t] = bv[t];
    __syncthreads();

    int g = blockIdx.x * 256 + t;          // 0 .. B*N-1
    int b = g >> 12;
    int n = g & (N_ - 1);

    float xc[C_];
    const float* xb = x + ((size_t)b * C_) * N_ + n;
#pragma unroll
    for (int c = 0; c < C_; ++c) xc[c] = xb[(size_t)c * N_];

    unsigned qw[4], kw[4];
#pragma unroll
    for (int p = 0; p < 4; ++p) {
        float a0q = sbq[2*p], a1q = sbq[2*p+1];
        float a0k = sbk[2*p], a1k = sbk[2*p+1];
#pragma unroll
        for (int c = 0; c < C_; ++c) {
            a0q = fmaf(swq[(2*p    ) * C_ + c], xc[c], a0q);
            a1q = fmaf(swq[(2*p + 1) * C_ + c], xc[c], a1q);
            a0k = fmaf(swk[(2*p    ) * C_ + c], xc[c], a0k);
            a1k = fmaf(swk[(2*p + 1) * C_ + c], xc[c], a1k);
        }
        qw[p] = (unsigned)f2bf(a0q) | ((unsigned)f2bf(a1q) << 16);
        kw[p] = (unsigned)f2bf(a0k) | ((unsigned)f2bf(a1k) << 16);
    }
    const uint4 z4 = make_uint4(0u, 0u, 0u, 0u);
    Qp[(size_t)g * 2]     = make_uint4(qw[0], qw[1], qw[2], qw[3]);
    Qp[(size_t)g * 2 + 1] = z4;                       // K = 8..15 zero padding
    Kp[(size_t)g * 2]     = make_uint4(kw[0], kw[1], kw[2], kw[3]);
    Kp[(size_t)g * 2 + 1] = z4;

    for (int o = 0; o < C_; ++o) {
        float a = sbv[o];
#pragma unroll
        for (int c = 0; c < C_; ++c) a = fmaf(swv[o * C_ + c], xc[c], a);
        Vb[((size_t)b * C_ + o) * N_ + n] = f2bf(a);
    }
}

// ---------------------------------------------------------------------------
// Kernel 2: segmented per-row softmax statistics.  One thread owns row n for
// one j-segment (4 segments -> 4x parallelism); K staged in LDS; online
// max/sum.  Partial (m, d) per segment.
// ---------------------------------------------------------------------------
__global__ __launch_bounds__(256) void stats_kernel(
    const uint4* __restrict__ Qp, const uint4* __restrict__ Kp,
    float* __restrict__ Mpart, float* __restrict__ Dpart)
{
    __shared__ unsigned sk[CH_ * 4];
    int t   = threadIdx.x;
    int rb  = blockIdx.x & 63;         // 64 row-blocks
    int seg = blockIdx.x >> 6;         // 4 segments
    int g   = rb * 256 + t;            // row id 0 .. B*N-1
    int b   = g >> 12;

    uint4 q = Qp[(size_t)g * 2];
    float qf[8];
    qf[0] = bf2f((unsigned short)(q.x & 0xFFFF)); qf[1] = bf2f((unsigned short)(q.x >> 16));
    qf[2] = bf2f((unsigned short)(q.y & 0xFFFF)); qf[3] = bf2f((unsigned short)(q.y >> 16));
    qf[4] = bf2f((unsigned short)(q.z & 0xFFFF)); qf[5] = bf2f((unsigned short)(q.z >> 16));
    qf[6] = bf2f((unsigned short)(q.w & 0xFFFF)); qf[7] = bf2f((unsigned short)(q.w >> 16));

    float m = -3.0e38f, d = 0.0f;
    for (int j0 = seg * SEGJ_; j0 < (seg + 1) * SEGJ_; j0 += CH_) {
        __syncthreads();
        uint4 kv = Kp[((size_t)b * N_ + j0 + t) * 2];
        sk[t * 4 + 0] = kv.x; sk[t * 4 + 1] = kv.y;
        sk[t * 4 + 2] = kv.z; sk[t * 4 + 3] = kv.w;
        __syncthreads();
        for (int jj = 0; jj < CH_; ++jj) {
            float s = 0.0f;
#pragma unroll
            for (int w = 0; w < 4; ++w) {
                unsigned kk = sk[jj * 4 + w];
                s = fmaf(qf[2*w],     bf2f((unsigned short)(kk & 0xFFFF)), s);
                s = fmaf(qf[2*w + 1], bf2f((unsigned short)(kk >> 16)),   s);
            }
            float mn = fmaxf(m, s);
            d = d * __expf(m - mn) + __expf(s - mn);
            m = mn;
        }
    }
    Mpart[seg * (B_ * N_) + g] = m;
    Dpart[seg * (B_ * N_) + g] = d;
}

// ---------------------------------------------------------------------------
// Kernel 2b: merge segment partials; emit fused softmax constant in the exp2
// domain:  Crow[n] = -log2(D[n]) - M[n]*log2e,
// so that  P[n,m] = exp2(S[n,m]*log2e + Crow[n]).
// ---------------------------------------------------------------------------
__global__ __launch_bounds__(256) void crow_kernel(
    const float* __restrict__ Mpart, const float* __restrict__ Dpart,
    float* __restrict__ Crow)
{
    int g = blockIdx.x * 256 + threadIdx.x;
    float m = -3.0e38f;
#pragma unroll
    for (int s = 0; s < NSEG_; ++s) m = fmaxf(m, Mpart[s * (B_ * N_) + g]);
    float d = 0.0f;
#pragma unroll
    for (int s = 0; s < NSEG_; ++s)
        d += Dpart[s * (B_ * N_) + g] *
             __builtin_amdgcn_exp2f((Mpart[s * (B_ * N_) + g] - m) * LOG2E_);
    Crow[g] = -__builtin_amdgcn_logf(d) - m * LOG2E_;   // v_log_f32 is log2
}

// ---------------------------------------------------------------------------
// Kernel 3: out = V @ softmax_rows(QtK) + x, one 16-col m-tile per wave.
// Key identity: the WMMA C-layout of the two S tiles (lane holds rows
// 8*half+r / 16+8*half+r of column l%16) is exactly the lane-local data the
// PV B-operand layout needs (K = 2j+8*half pairs of column l%16), so the
// softmaxed P tile is repacked C->B entirely in registers: fma+exp2 per
// element, one v_perm_b32 per bf16 pair.  No LDS, no divergence (Q/K rows
// carry their own zero padding).  6 WMMAs per 32-n chunk.
// ---------------------------------------------------------------------------
__global__ __launch_bounds__(128) void attn_out_kernel(
    const uint4* __restrict__ Qp, const uint4* __restrict__ Kp,
    const unsigned short* __restrict__ Vb,
    const float* __restrict__ Crow,
    const float* __restrict__ x, float* __restrict__ out)
{
    int t    = threadIdx.x;
    int wave = t >> 5, lane = t & 31;
    int half = lane >> 4, col = lane & 15;
    int mtile = blockIdx.x * 4 + wave;             // 0 .. B*N/16-1
    int b  = mtile >> 8;
    int m0 = (mtile & 255) << 4;

    const size_t bN = (size_t)b * N_;

    // B-operand for scores: K columns m0..m0+15 (fixed for the whole wave).
    // Unconditional load: upper-half lanes read the stored zero padding.
    V16 kb;
    {
        uint4 kv = Kp[(bN + m0 + col) * 2 + half];
        kb.u[0] = kv.x; kb.u[1] = kv.y; kb.u[2] = kv.z; kb.u[3] = kv.w;
        kb.u[4] = 0u;   kb.u[5] = 0u;   kb.u[6] = 0u;   kb.u[7] = 0u;
    }

    v8f acc0 = {0.f,0.f,0.f,0.f,0.f,0.f,0.f,0.f};
    v8f acc1 = acc0, acc2 = acc0, acc3 = acc0;
    const v8f zc = acc0;
    const unsigned short* vbase = Vb + ((size_t)b * C_) * N_;

    for (int n0 = 0; n0 < N_; n0 += 32) {
        // A-operands: Q rows n0..n0+31 (two 16-row tiles), unconditional.
        V16 aq0, aq1;
        {
            uint4 v0 = Qp[(bN + n0 + col) * 2 + half];
            uint4 v1 = Qp[(bN + n0 + 16 + col) * 2 + half];
            aq0.u[0] = v0.x; aq0.u[1] = v0.y; aq0.u[2] = v0.z; aq0.u[3] = v0.w;
            aq1.u[0] = v1.x; aq1.u[1] = v1.y; aq1.u[2] = v1.z; aq1.u[3] = v1.w;
            aq0.u[4] = aq0.u[5] = aq0.u[6] = aq0.u[7] = 0u;
            aq1.u[4] = aq1.u[5] = aq1.u[6] = aq1.u[7] = 0u;
        }
        v8f s0 = __builtin_amdgcn_wmma_f32_16x16x32_bf16(false, aq0.v, false, kb.v,
                                                         (short)0, zc, false, false);
        v8f s1 = __builtin_amdgcn_wmma_f32_16x16x32_bf16(false, aq1.v, false, kb.v,
                                                         (short)0, zc, false, false);

        // P = exp2(s*log2e + Crow[row]); C-layout -> B-layout is lane-local:
        // pb pair j<4 = rows {8h+2j, 8h+2j+1} of tile 0, pair j>=4 = tile 1.
        V16 pb;
#pragma unroll
        for (int tp = 0; tp < 4; ++tp) {
            int r0   = 2 * tp;
            int row0 = 8 * half + r0;              // tile 0 rows
            float p0 = __builtin_amdgcn_exp2f(fmaf(s0[r0],     LOG2E_, Crow[bN + n0 + row0]));
            float p1 = __builtin_amdgcn_exp2f(fmaf(s0[r0 + 1], LOG2E_, Crow[bN + n0 + row0 + 1]));
            pb.u[tp] = __builtin_amdgcn_perm(__float_as_uint(p1), __float_as_uint(p0), 0x07060302u);
            int row1 = 16 + row0;                  // tile 1 rows
            float p2 = __builtin_amdgcn_exp2f(fmaf(s1[r0],     LOG2E_, Crow[bN + n0 + row1]));
            float p3 = __builtin_amdgcn_exp2f(fmaf(s1[r0 + 1], LOG2E_, Crow[bN + n0 + row1 + 1]));
            pb.u[4 + tp] = __builtin_amdgcn_perm(__float_as_uint(p3), __float_as_uint(p2), 0x07060302u);
        }

        // 4 c-tiles of V as A-operands (two B128 loads each); accumulate.
#pragma unroll
        for (int ct = 0; ct < 4; ++ct) {
            const unsigned short* vrow = vbase + (size_t)(ct * 16 + col) * N_;
            uint4 a0 = *(const uint4*)(vrow + n0 + 8 * half);
            uint4 a1 = *(const uint4*)(vrow + n0 + 16 + 8 * half);
            V16 av;
            av.u[0] = a0.x; av.u[1] = a0.y; av.u[2] = a0.z; av.u[3] = a0.w;
            av.u[4] = a1.x; av.u[5] = a1.y; av.u[6] = a1.z; av.u[7] = a1.w;
            if (ct == 0) acc0 = __builtin_amdgcn_wmma_f32_16x16x32_bf16(false, av.v, false, pb.v, (short)0, acc0, false, false);
            if (ct == 1) acc1 = __builtin_amdgcn_wmma_f32_16x16x32_bf16(false, av.v, false, pb.v, (short)0, acc1, false, false);
            if (ct == 2) acc2 = __builtin_amdgcn_wmma_f32_16x16x32_bf16(false, av.v, false, pb.v, (short)0, acc2, false, false);
            if (ct == 3) acc3 = __builtin_amdgcn_wmma_f32_16x16x32_bf16(false, av.v, false, pb.v, (short)0, acc3, false, false);
        }
    }

    // Epilogue: fused residual  out = acc + x
#pragma unroll
    for (int ct = 0; ct < 4; ++ct) {
        v8f a = ct == 0 ? acc0 : ct == 1 ? acc1 : ct == 2 ? acc2 : acc3;
#pragma unroll
        for (int r = 0; r < 8; ++r) {
            int c = ct * 16 + 8 * half + r;
            size_t idx = ((size_t)b * C_ + c) * N_ + m0 + col;
            out[idx] = a[r] + x[idx];
        }
    }
}

// ---------------------------------------------------------------------------
extern "C" void kernel_launch(void* const* d_in, const int* in_sizes, int n_in,
                              void* d_out, int out_size, void* d_ws, size_t ws_size,
                              hipStream_t stream)
{
    const float* x  = (const float*)d_in[0];
    const float* wq = (const float*)d_in[1];
    const float* bq = (const float*)d_in[2];
    const float* wk = (const float*)d_in[3];
    const float* bk = (const float*)d_in[4];
    const float* wv = (const float*)d_in[5];
    const float* bv = (const float*)d_in[6];
    float* out = (float*)d_out;

    char* ws = (char*)d_ws;
    const size_t qp_bytes = (size_t)B_ * N_ * 16 * 2;           // 512 KB (padded rows)
    const size_t vb_bytes = (size_t)B_ * C_ * N_ * 2;           // 2 MB
    const size_t md_bytes = (size_t)NSEG_ * B_ * N_ * 4;        // 256 KB

    uint4*          Qp    = (uint4*)(ws);
    uint4*          Kp    = (uint4*)(ws + qp_bytes);
    unsigned short* Vb    = (unsigned short*)(ws + 2 * qp_bytes);
    float*          Mpart = (float*)(ws + 2 * qp_bytes + vb_bytes);
    float*          Dpart = (float*)(ws + 2 * qp_bytes + vb_bytes + md_bytes);
    float*          Crow  = (float*)(ws + 2 * qp_bytes + vb_bytes + 2 * md_bytes);

    proj_kernel<<<(B_ * N_) / 256, 256, 0, stream>>>(x, wq, bq, wk, bk, wv, bv, Qp, Kp, Vb);
    stats_kernel<<<NSEG_ * (B_ * N_) / 256, 256, 0, stream>>>(Qp, Kp, Mpart, Dpart);
    crow_kernel<<<(B_ * N_) / 256, 256, 0, stream>>>(Mpart, Dpart, Crow);
    attn_out_kernel<<<(B_ * (N_ / 16)) / 4, 128, 0, stream>>>(Qp, Kp, Vb, Crow, x, out);
}
